// AdaLNZeroBlock_39651138076997
// MI455X (gfx1250) — matrix operand order repaired and checked by reading, not compiled, not run on hardware
//
#include <hip/hip_runtime.h>
#include <hip/hip_bf16.h>
#include <stdint.h>

// ---------------------------------------------------------------------------
// Problem constants
// ---------------------------------------------------------------------------
#define BB     2
#define SS     2048
#define DD     1024
#define HH     16
#define DHD    64
#define INNER  2730
#define TWOI   5460          // 2*INNER
#define NPAD_FI 5504         // 2*INNER padded to multiple of 128
#define KPAD_FO 2752         // INNER padded to multiple of 32
#define MROWS  (BB*SS)       // 4096
#define SIXD   (6*DD)        // 6144

typedef __attribute__((ext_vector_type(16))) __bf16 v16bf;
typedef __attribute__((ext_vector_type(8)))  float  v8f;
typedef unsigned int u32x4 __attribute__((ext_vector_type(4)));
typedef int          i32x8 __attribute__((ext_vector_type(8)));
typedef int          i32x4 __attribute__((ext_vector_type(4)));

#if defined(__has_builtin)
#if __has_builtin(__builtin_amdgcn_tensor_load_to_lds)
#define HAVE_TDM 1
#endif
#endif
#ifndef HAVE_TDM
#define HAVE_TDM 0
#endif

__device__ __forceinline__ float  bf2f(__bf16 x) { return (float)x; }
__device__ __forceinline__ __bf16 f2bf(float x)  { return (__bf16)x; }

// 16-bit A-matrix 16x32 layout (ISA 7.12.2): lane m = lane&15, g = lane>>4.
__device__ __forceinline__ int kbaseA(int v, int g) {
    return (v < 4) ? (v * 2 + g * 8) : (16 + (v - 4) * 2 + g * 8);
}
// 16-bit B-matrix 32x16 layout: same K order as A (lanes 0-15 K=0..7,16..23; lanes 16-31 K=8..15,24..31).
__device__ __forceinline__ int kbaseB(int v, int g) {
    return (v < 4) ? (v * 2 + g * 8) : (16 + (v - 4) * 2 + g * 8);
}

union FragBF {
    v16bf    bf;
    uint32_t u[8];
};

__device__ __forceinline__ v8f wmma_bf16(v16bf a, v16bf b, v8f c) {
    return __builtin_amdgcn_wmma_f32_16x16x32_bf16(
        false, a, false, b, (short)0, c, false, false);
}

// ds_swizzle xor-butterfly within a 32-lane group-of-32 pattern
template <int XMASK>
__device__ __forceinline__ float swz_xor(float v) {
    return __int_as_float(__builtin_amdgcn_ds_swizzle(
        __float_as_int(v), (XMASK << 10) | 0x1F));
}

// ---------------------------------------------------------------------------
// Tensor Data Mover: issue a 2D tile load (tile1 rows x tile0 bf16 elems,
// row stride `strideElems`) from global into LDS at `ldsPtr`.
// D# layout per CDNA5 ISA 8.3/8.4 (count=1, type=2, data_size=2B).
// Toolchain uses the 6-arg builtin: (g0, g1, g2, g3, g_extra, cpol).
// ---------------------------------------------------------------------------
__device__ __forceinline__ void tdm_issue_2d(const void* gptr, const void* ldsPtr,
                                             int tile0, int tile1,
                                             long long strideElems)
{
#if HAVE_TDM
    uint64_t ga = (uint64_t)(uintptr_t)gptr;
    unsigned la = (unsigned)(uintptr_t)ldsPtr;     // low 32 bits = LDS offset
    u32x4 g0;
    g0[0] = 1u;                                    // count=1, user mode
    g0[1] = la;                                    // lds_addr
    g0[2] = (unsigned)ga;                          // global_addr[31:0]
    g0[3] = (unsigned)((ga >> 32) & 0x01FFFFFFu) | 0x80000000u; // addr[56:32] | type=2
    unsigned t0 = (unsigned)tile0, t1 = (unsigned)tile1;
    uint64_t st = (uint64_t)strideElems;
    i32x8 g1;
    g1[0] = (int)(1u << 16);                       // data_size=1 (2 bytes)
    g1[1] = (int)((t0 & 0xFFFFu) << 16);           // tensor_dim0[15:0] (== tile0)
    g1[2] = (int)((t0 >> 16) | ((t1 & 0xFFFFu) << 16)); // tdim0 hi | tensor_dim1 lo
    g1[3] = (int)((t1 >> 16) | (t0 << 16));        // tdim1 hi | tile_dim0
    g1[4] = (int)(t1 & 0xFFFFu);                   // tile_dim1 (tile_dim2 = 0)
    g1[5] = (int)(unsigned)(st & 0xFFFFFFFFu);     // tensor_dim0_stride lo32
    g1[6] = (int)(unsigned)((st >> 32) & 0xFFFFu); // stride hi16 (dim1_stride=0)
    g1[7] = 0;
    i32x4 z4 = {0, 0, 0, 0};
    i32x8 z8 = {0, 0, 0, 0, 0, 0, 0, 0};
    __builtin_amdgcn_tensor_load_to_lds(g0, g1, z4, z4, z8, 0);
#else
    (void)gptr; (void)ldsPtr; (void)tile0; (void)tile1; (void)strideElems;
#endif
}

__device__ __forceinline__ void tdm_wait0() {
#if HAVE_TDM
    __builtin_amdgcn_s_wait_tensorcnt((short)0);
#endif
}

// ---------------------------------------------------------------------------
// Weight transpose + f32 -> bf16 convert (with zero padding)
// ---------------------------------------------------------------------------
__global__ __launch_bounds__(256) void convT_kernel(
    const float* __restrict__ src, __bf16* __restrict__ dst,
    int K, int N, int Kpad, long total)
{
    long idx = (long)blockIdx.x * 256 + threadIdx.x;
    if (idx >= total) return;
    int n = (int)(idx / Kpad);
    int k = (int)(idx - (long)n * Kpad);
    float v = (n < N && k < K) ? src[(size_t)k * N + n] : 0.f;
    dst[idx] = f2bf(v);
}

// ---------------------------------------------------------------------------
// mod = silu(cond) @ w_mod + b_mod       (B x 6D), tiny
// ---------------------------------------------------------------------------
__global__ __launch_bounds__(256) void mod_kernel(
    const float* __restrict__ cond, const float* __restrict__ w,
    const float* __restrict__ bias, float* __restrict__ mod)
{
    int idx = blockIdx.x * 256 + threadIdx.x;
    if (idx >= BB * SIXD) return;
    int b = idx / SIXD;
    int c = idx - b * SIXD;
    float acc = bias[c];
    for (int k = 0; k < DD; ++k) {
        float cv = cond[b * DD + k];
        float s  = cv / (1.f + __expf(-cv));
        acc = fmaf(s, w[(size_t)k * SIXD + c], acc);
    }
    mod[idx] = acc;
}

// ---------------------------------------------------------------------------
// LayerNorm + modulate -> bf16
// ---------------------------------------------------------------------------
__global__ __launch_bounds__(256) void ln_mod_kernel(
    const float* __restrict__ x, const float* __restrict__ mod,
    int shOff, int scOff, __bf16* __restrict__ out)
{
    int row = blockIdx.x;
    int b   = row >> 11;
    int tid = threadIdx.x;
    __shared__ float s1[256], s2[256];
    float vals[4];
    float a = 0.f, q = 0.f;
    #pragma unroll
    for (int i = 0; i < 4; ++i) {
        float v = x[(size_t)row * DD + i * 256 + tid];
        vals[i] = v; a += v; q += v * v;
    }
    s1[tid] = a; s2[tid] = q;
    __syncthreads();
    for (int off = 128; off > 0; off >>= 1) {
        if (tid < off) { s1[tid] += s1[tid + off]; s2[tid] += s2[tid + off]; }
        __syncthreads();
    }
    float mean = s1[0] * (1.f / DD);
    float var  = s2[0] * (1.f / DD) - mean * mean;
    float rs   = rsqrtf(var + 1e-5f);
    #pragma unroll
    for (int i = 0; i < 4; ++i) {
        int c = i * 256 + tid;
        float sh  = mod[(size_t)b * SIXD + shOff + c];
        float scl = mod[(size_t)b * SIXD + scOff + c];
        out[(size_t)row * DD + c] = f2bf((vals[i] - mean) * rs * (1.f + scl) + sh);
    }
}

// ---------------------------------------------------------------------------
// bf16 WMMA GEMM with TDM double-buffered LDS staging.
// C[M,N] = A[M,K] * Bt[N,K]^T (+ bias).  Block tile 128x128, 8 waves (4x2),
// wave tile 32x64.  MODE 0: f32 out, 1: bf16 out, 2: resid + gate * acc.
// ---------------------------------------------------------------------------
template <int MODE>
__global__ __launch_bounds__(256, 1) void gemm_bf16_kernel(
    const __bf16* __restrict__ A, const __bf16* __restrict__ Bt,
    const float* __restrict__ bias, int Ntrue,
    float* __restrict__ Cf, __bf16* __restrict__ Cb,
    const float* __restrict__ resid, const float* __restrict__ gatev,
    int gstride, int rowsPerBatch, float* __restrict__ outf,
    int K, int lda, int ldb, int ldc)
{
    __shared__ __bf16 ldsA[2][128 * 32];
    __shared__ __bf16 ldsB[2][128 * 32];

    const int lane = threadIdx.x & 31;
    const int wv   = threadIdx.x >> 5;
    const int wm   = wv & 3;
    const int wn   = wv >> 2;
    const int ml   = lane & 15;
    const int g    = lane >> 4;
    const int mBlk = blockIdx.y * 128;
    const int nBlk = blockIdx.x * 128;

    v8f zero8 = {0.f, 0.f, 0.f, 0.f, 0.f, 0.f, 0.f, 0.f};
    v8f acc[2][4];
    #pragma unroll
    for (int i = 0; i < 2; ++i)
        #pragma unroll
        for (int j = 0; j < 4; ++j) acc[i][j] = zero8;

    const int T = K >> 5;

#if HAVE_TDM
    if (threadIdx.x < 32) {
        tdm_issue_2d(A  + (size_t)mBlk * lda, &ldsA[0][0], 32, 128, lda);
        tdm_issue_2d(Bt + (size_t)nBlk * ldb, &ldsB[0][0], 32, 128, ldb);
    }
#endif

    for (int i = 0; i < T; ++i) {
        const int cur = i & 1;
        const int kk  = i << 5;
#if HAVE_TDM
        if (threadIdx.x < 32) tdm_wait0();
        __syncthreads();
        if (i + 1 < T && threadIdx.x < 32) {
            tdm_issue_2d(A  + (size_t)mBlk * lda + kk + 32, &ldsA[1 - cur][0], 32, 128, lda);
            tdm_issue_2d(Bt + (size_t)nBlk * ldb + kk + 32, &ldsB[1 - cur][0], 32, 128, ldb);
        }
#else
        __syncthreads();
        for (int t = threadIdx.x; t < 128 * 16; t += 256) {
            int r = t >> 4, c = (t & 15) << 1;
            *(uint32_t*)&ldsA[cur][r * 32 + c] =
                *(const uint32_t*)(A + (size_t)(mBlk + r) * lda + kk + c);
            *(uint32_t*)&ldsB[cur][r * 32 + c] =
                *(const uint32_t*)(Bt + (size_t)(nBlk + r) * ldb + kk + c);
        }
        __syncthreads();
#endif
        FragBF a[2];
        #pragma unroll
        for (int i2 = 0; i2 < 2; ++i2) {
            const __bf16* ap = &ldsA[cur][(wm * 32 + i2 * 16 + ml) * 32];
            #pragma unroll
            for (int v = 0; v < 8; ++v)
                a[i2].u[v] = *(const uint32_t*)(ap + kbaseA(v, g));
        }
        FragBF bfr[4];
        #pragma unroll
        for (int j = 0; j < 4; ++j) {
            const __bf16* bp = &ldsB[cur][(wn * 64 + j * 16 + ml) * 32];
            #pragma unroll
            for (int v = 0; v < 8; ++v)
                bfr[j].u[v] = *(const uint32_t*)(bp + kbaseB(v, g));
        }
        #pragma unroll
        for (int i2 = 0; i2 < 2; ++i2)
            #pragma unroll
            for (int j = 0; j < 4; ++j)
                acc[i2][j] = wmma_bf16(a[i2].bf, bfr[j].bf, acc[i2][j]);
    }

    // Epilogue.  C layout: VGPR r -> row (r + 8g), lane&15 -> col.
    #pragma unroll
    for (int i = 0; i < 2; ++i) {
        #pragma unroll
        for (int j = 0; j < 4; ++j) {
            #pragma unroll
            for (int r = 0; r < 8; ++r) {
                int row = mBlk + wm * 32 + i * 16 + r + g * 8;
                int col = nBlk + wn * 64 + j * 16 + ml;
                float v = acc[i][j][r];
                v += (col < Ntrue) ? bias[col] : 0.f;
                if (MODE == 0) {
                    Cf[(size_t)row * ldc + col] = v;
                } else if (MODE == 1) {
                    Cb[(size_t)row * ldc + col] = f2bf(v);
                } else {
                    int b = row / rowsPerBatch;
                    outf[(size_t)row * ldc + col] =
                        resid[(size_t)row * ldc + col] +
                        gatev[(size_t)b * gstride + col] * v;
                }
            }
        }
    }
}

// ---------------------------------------------------------------------------
// RoPE + L2-norm + layout shuffle (scale DH^-0.5 folded into q).
// qb,kb: (B*H, S, 64) bf16.   vT: (B*H, 64, S) bf16 (transposed).
// ---------------------------------------------------------------------------
__global__ __launch_bounds__(64) void rope_kernel(
    const float* __restrict__ qkv, __bf16* __restrict__ qb,
    __bf16* __restrict__ kb, __bf16* __restrict__ vT)
{
    int idx = blockIdx.x;            // (b*H+h)*S + s
    int s   = idx & (SS - 1);
    int bh  = idx >> 11;
    int b   = bh >> 4;
    int h   = bh & 15;
    int d   = threadIdx.x;

    size_t rowbase = ((size_t)(b * SS + s)) * (3 * DD);
    float qv = qkv[rowbase + 0 * DD + h * DHD + d];
    float kv = qkv[rowbase + 1 * DD + h * DHD + d];
    float vv = qkv[rowbase + 2 * DD + h * DHD + d];

    int   i    = d >> 1;
    float freq = __expf(-(float)i * (9.21034037198e0f / 32.f)); // ln(1e4)/32
    float th   = (float)s * freq;
    float sn = __sinf(th), cs = __cosf(th);
    float qo = __shfl_xor(qv, 1);
    float ko = __shfl_xor(kv, 1);
    float qr, kr;
    if ((d & 1) == 0) { qr = qv * cs - qo * sn; kr = kv * cs - ko * sn; }
    else              { qr = qo * sn + qv * cs; kr = ko * sn + kv * cs; }

    __shared__ float rq[64], rk[64];
    rq[d] = qr * qr; rk[d] = kr * kr;
    __syncthreads();
    for (int off = 32; off > 0; off >>= 1) {
        if (d < off) { rq[d] += rq[d + off]; rk[d] += rk[d + off]; }
        __syncthreads();
    }
    float qn = qr / fmaxf(sqrtf(rq[0]), 1e-12f) * 0.125f;
    float kn = kr / fmaxf(sqrtf(rk[0]), 1e-12f);

    qb[((size_t)bh * SS + s) * DHD + d] = f2bf(qn);
    kb[((size_t)bh * SS + s) * DHD + d] = f2bf(kn);
    vT[((size_t)bh * DHD + d) * SS + s] = f2bf(vv);
}

// ---------------------------------------------------------------------------
// Flash attention with TDM-staged K/V tiles shared by the 4 waves.
// Block = 4 waves, each wave owns one 16-row q tile; 32-key steps with
// online softmax.  grid = (S/64, B*H)
// ---------------------------------------------------------------------------
__global__ __launch_bounds__(128, 1) void attn_kernel(
    const __bf16* __restrict__ qb, const __bf16* __restrict__ kb,
    const __bf16* __restrict__ vT, __bf16* __restrict__ obuf)
{
    __shared__ __bf16 ldsK[2][32 * 64];   // keys x dh
    __shared__ __bf16 ldsV[2][64 * 32];   // dh x keys
    __shared__ __bf16 pLds[4][16 * 32];

    const int w    = threadIdx.x >> 5;
    const int lane = threadIdx.x & 31;
    const int ml   = lane & 15;
    const int g    = lane >> 4;
    const int bh   = blockIdx.y;
    const int b    = bh >> 4;
    const int h    = bh & 15;
    const int s0   = blockIdx.x * 64 + w * 16;

    const size_t hbS = (size_t)bh * SS;
    const size_t hbD = (size_t)bh * DHD;

    FragBF aq[2];
    #pragma unroll
    for (int t = 0; t < 2; ++t) {
        const __bf16* qp = qb + (hbS + s0 + ml) * DHD + t * 32;
        #pragma unroll
        for (int v = 0; v < 8; ++v)
            aq[t].u[v] = *(const uint32_t*)(qp + kbaseA(v, g));
    }

    v8f zero8 = {0.f, 0.f, 0.f, 0.f, 0.f, 0.f, 0.f, 0.f};
    v8f oacc[4];
    #pragma unroll
    for (int nt = 0; nt < 4; ++nt) oacc[nt] = zero8;
    float mr[8], lr[8];
    #pragma unroll
    for (int r = 0; r < 8; ++r) { mr[r] = -3.0e38f; lr[r] = 0.f; }

    const int T = SS / 32;
#if HAVE_TDM
    if (threadIdx.x < 32) {
        tdm_issue_2d(kb + hbS * DHD, &ldsK[0][0], 64, 32, DHD);
        tdm_issue_2d(vT + hbD * SS,  &ldsV[0][0], 32, 64, SS);
    }
#endif

    for (int i = 0; i < T; ++i) {
        const int cur = i & 1;
        const int kt  = i << 5;
#if HAVE_TDM
        if (threadIdx.x < 32) tdm_wait0();
        __syncthreads();
        if (i + 1 < T && threadIdx.x < 32) {
            tdm_issue_2d(kb + (hbS + kt + 32) * DHD, &ldsK[1 - cur][0], 64, 32, DHD);
            tdm_issue_2d(vT + hbD * SS + kt + 32,    &ldsV[1 - cur][0], 32, 64, SS);
        }
#else
        __syncthreads();
        for (int t = threadIdx.x; t < 1024; t += 128) {
            int r = t >> 5, c = (t & 31) << 1;
            *(uint32_t*)&ldsK[cur][r * 64 + c] =
                *(const uint32_t*)(kb + (hbS + kt + r) * DHD + c);
        }
        for (int t = threadIdx.x; t < 1024; t += 128) {
            int r = t >> 4, c = (t & 15) << 1;
            *(uint32_t*)&ldsV[cur][r * 32 + c] =
                *(const uint32_t*)(vT + (hbD + r) * SS + kt + c);
        }
        __syncthreads();
#endif
        // Scores: 16 q-rows x 32 keys
        v8f sc[2];
        #pragma unroll
        for (int j = 0; j < 2; ++j) {
            v8f s = zero8;
            #pragma unroll
            for (int t = 0; t < 2; ++t) {
                FragBF bk;
                const __bf16* kp = &ldsK[cur][(j * 16 + ml) * 64 + t * 32];
                #pragma unroll
                for (int v = 0; v < 8; ++v)
                    bk.u[v] = *(const uint32_t*)(kp + kbaseB(v, g));
                s = wmma_bf16(aq[t].bf, bk.bf, s);
            }
            sc[j] = s;
        }

        // Online softmax; rows of C live in 16-lane groups -> xor<16 stays local.
        #pragma unroll
        for (int r = 0; r < 8; ++r) {
            float s0v = sc[0][r];
            float s1v = sc[1][r];
            float rmax = fmaxf(s0v, s1v);
            rmax = fmaxf(rmax, swz_xor<1>(rmax));
            rmax = fmaxf(rmax, swz_xor<2>(rmax));
            rmax = fmaxf(rmax, swz_xor<4>(rmax));
            rmax = fmaxf(rmax, swz_xor<8>(rmax));
            float mnew  = fmaxf(mr[r], rmax);
            float alpha = __expf(mr[r] - mnew);
            float p0 = __expf(s0v - mnew);
            float p1 = __expf(s1v - mnew);
            float rsum = p0 + p1;
            rsum += swz_xor<1>(rsum);
            rsum += swz_xor<2>(rsum);
            rsum += swz_xor<4>(rsum);
            rsum += swz_xor<8>(rsum);
            lr[r] = lr[r] * alpha + rsum;
            mr[r] = mnew;
            #pragma unroll
            for (int nt = 0; nt < 4; ++nt) oacc[nt][r] *= alpha;
            int prow = r + g * 8;
            pLds[w][prow * 32 + ml]      = f2bf(p0);
            pLds[w][prow * 32 + 16 + ml] = f2bf(p1);
        }
        asm volatile("s_wait_dscnt 0x0" ::: "memory");

        // Re-form P as an A-fragment (16x32 over keys) from LDS.
        FragBF ap;
        #pragma unroll
        for (int v = 0; v < 8; ++v)
            ap.u[v] = *(const uint32_t*)(&pLds[w][ml * 32 + kbaseA(v, g)]);

        // O += P * V  (N = 64 dh in 4 tiles, K = 32 keys)
        #pragma unroll
        for (int nt = 0; nt < 4; ++nt) {
            FragBF bv;
            const __bf16* vp = &ldsV[cur][(nt * 16 + ml) * 32];
            #pragma unroll
            for (int v = 0; v < 8; ++v)
                bv.u[v] = *(const uint32_t*)(vp + kbaseB(v, g));
            oacc[nt] = wmma_bf16(ap.bf, bv.bf, oacc[nt]);
        }
    }

    #pragma unroll
    for (int r = 0; r < 8; ++r) {
        float inv = 1.f / lr[r];
        int srow = s0 + r + g * 8;
        #pragma unroll
        for (int nt = 0; nt < 4; ++nt) {
            obuf[((size_t)b * SS + srow) * DD + h * DHD + nt * 16 + ml] =
                f2bf(oacc[nt][r] * inv);
        }
    }
}

// ---------------------------------------------------------------------------
// SwiGLU:  g[row, j] = value * silu(gate)
// ---------------------------------------------------------------------------
__global__ __launch_bounds__(256) void swiglu_kernel(
    const __bf16* __restrict__ h, __bf16* __restrict__ gout)
{
    long idx = (long)blockIdx.x * 256 + threadIdx.x;
    if (idx >= (long)MROWS * KPAD_FO) return;
    int row = (int)(idx / KPAD_FO);
    int j   = (int)(idx - (long)row * KPAD_FO);
    float r = 0.f;
    if (j < INNER) {
        float val = bf2f(h[(size_t)row * NPAD_FI + j]);
        float gt  = bf2f(h[(size_t)row * NPAD_FI + INNER + j]);
        r = val * (gt / (1.f + __expf(-gt)));
    }
    gout[idx] = f2bf(r);
}

// ---------------------------------------------------------------------------
// Host launch
// ---------------------------------------------------------------------------
extern "C" void kernel_launch(void* const* d_in, const int* in_sizes, int n_in,
                              void* d_out, int out_size, void* d_ws, size_t ws_size,
                              hipStream_t stream)
{
    (void)in_sizes; (void)n_in; (void)out_size; (void)ws_size;
    const float* tokens = (const float*)d_in[0];
    const float* cond   = (const float*)d_in[1];
    const float* w_qkv  = (const float*)d_in[2];
    const float* b_qkv  = (const float*)d_in[3];
    const float* w_ao   = (const float*)d_in[4];
    const float* b_ao   = (const float*)d_in[5];
    const float* w_fi   = (const float*)d_in[6];
    const float* b_fi   = (const float*)d_in[7];
    const float* w_fo   = (const float*)d_in[8];
    const float* b_fo   = (const float*)d_in[9];
    const float* w_mod  = (const float*)d_in[10];
    const float* b_mod  = (const float*)d_in[11];
    float* out = (float*)d_out;
    char* ws = (char*)d_ws;

    const size_t OFF_MOD   = 0;           //  49152  f32 (B,6D)
    const size_t OFF_XB    = 49152;       //  8 MB   bf16 x / reused as y
    const size_t OFF_WTQKV = 8437760;     //  6 MB   bf16 (3072,1024)
    const size_t OFF_WTAO  = 14729216;    //  2 MB   bf16 (1024,1024)
    const size_t OFF_WTFI  = 16826368;    // 11 MB   bf16 (5504,1024)
    const size_t OFF_WTFO  = 28098560;    //  5.6 MB bf16 (1024,2752)
    const size_t OFF_QKV   = 33734656;    // 48 MB   f32 qkv / reused as h bf16
    const size_t OFF_QB    = 84066304;    //  8 MB   bf16 / region reused as gbuf
    const size_t OFF_KB    = 92454912;    //  8 MB
    const size_t OFF_VT    = 100843520;   //  8 MB
    const size_t OFF_OBUF  = 109232128;   //  8 MB   bf16 O
    const size_t OFF_TOK2  = 117620736;   // 16 MB   f32 residual stream

    float*  mod   = (float*)(ws + OFF_MOD);
    __bf16* xb    = (__bf16*)(ws + OFF_XB);
    __bf16* wtqkv = (__bf16*)(ws + OFF_WTQKV);
    __bf16* wtao  = (__bf16*)(ws + OFF_WTAO);
    __bf16* wtfi  = (__bf16*)(ws + OFF_WTFI);
    __bf16* wtfo  = (__bf16*)(ws + OFF_WTFO);
    float*  qkv   = (float*)(ws + OFF_QKV);
    __bf16* hbuf  = (__bf16*)(ws + OFF_QKV);   // alias (qkv dead by then)
    __bf16* qb    = (__bf16*)(ws + OFF_QB);
    __bf16* kb    = (__bf16*)(ws + OFF_KB);
    __bf16* vT    = (__bf16*)(ws + OFF_VT);
    __bf16* gbuf  = (__bf16*)(ws + OFF_QB);    // alias (q/k/v dead by then)
    __bf16* obuf  = (__bf16*)(ws + OFF_OBUF);
    float*  tok2  = (float*)(ws + OFF_TOK2);

    // 1) Weight transposes -> bf16 (zero-padded)
    {
        long t;
        t = (long)3072 * 1024;
        convT_kernel<<<(int)((t + 255) / 256), 256, 0, stream>>>(w_qkv, wtqkv, 1024, 3072, 1024, t);
        t = (long)1024 * 1024;
        convT_kernel<<<(int)((t + 255) / 256), 256, 0, stream>>>(w_ao, wtao, 1024, 1024, 1024, t);
        t = (long)NPAD_FI * 1024;
        convT_kernel<<<(int)((t + 255) / 256), 256, 0, stream>>>(w_fi, wtfi, 1024, TWOI, 1024, t);
        t = (long)1024 * KPAD_FO;
        convT_kernel<<<(int)((t + 255) / 256), 256, 0, stream>>>(w_fo, wtfo, INNER, 1024, KPAD_FO, t);
    }

    // 2) mod = silu(cond) @ w_mod + b_mod
    mod_kernel<<<(BB * SIXD + 255) / 256, 256, 0, stream>>>(cond, w_mod, b_mod, mod);

    // 3) x = modulate(LN(tokens), shift_a, scale_a) -> bf16
    ln_mod_kernel<<<MROWS, 256, 0, stream>>>(tokens, mod, 0 * DD, 1 * DD, xb);

    // 4) qkv = x @ w_qkv + b_qkv            (MODE 0, f32 out)
    gemm_bf16_kernel<0><<<dim3(3072 / 128, MROWS / 128), 256, 0, stream>>>(
        xb, wtqkv, b_qkv, 3072, qkv, nullptr, nullptr, nullptr, 0, 0, nullptr,
        1024, 1024, 1024, 3072);

    // 5) RoPE + L2norm, split/transposed layouts
    rope_kernel<<<BB * HH * SS, 64, 0, stream>>>(qkv, qb, kb, vT);

    // 6) flash attention -> obuf (B,S,D) bf16
    attn_kernel<<<dim3(SS / 64, BB * HH), 128, 0, stream>>>(qb, kb, vT, obuf);

    // 7) tok2 = tokens + gate_a * (O @ w_attn_out + b)   (MODE 2)
    gemm_bf16_kernel<2><<<dim3(1024 / 128, MROWS / 128), 256, 0, stream>>>(
        obuf, wtao, b_ao, 1024, nullptr, nullptr, tokens, mod + 2 * DD,
        SIXD, SS, tok2, 1024, 1024, 1024, 1024);

    // 8) y = modulate(LN(tok2), shift_m, scale_m) -> bf16 (reuse xb)
    ln_mod_kernel<<<MROWS, 256, 0, stream>>>(tok2, mod, 3 * DD, 4 * DD, xb);

    // 9) h = y @ w_ffn_in + b_ffn_in        (MODE 1, bf16 out, padded N)
    gemm_bf16_kernel<1><<<dim3(NPAD_FI / 128, MROWS / 128), 256, 0, stream>>>(
        xb, wtfi, b_fi, TWOI, nullptr, hbuf, nullptr, nullptr, 0, 0, nullptr,
        1024, 1024, 1024, NPAD_FI);

    // 10) g = value * silu(gate)
    {
        long t = (long)MROWS * KPAD_FO;
        swiglu_kernel<<<(int)((t + 255) / 256), 256, 0, stream>>>(hbuf, gbuf);
    }

    // 11) out = tok2 + gate_m * (g @ w_ffn_out + b)   (MODE 2)
    gemm_bf16_kernel<2><<<dim3(1024 / 128, MROWS / 128), 256, 0, stream>>>(
        gbuf, wtfo, b_fo, 1024, nullptr, nullptr, tok2, mod + 5 * DD,
        SIXD, SS, out, KPAD_FO, KPAD_FO, KPAD_FO, 1024);
}
